// My_vgg16_74990128988339
// MI455X (gfx1250) — matrix operand order repaired and checked by reading, not compiled, not run on hardware
//
#include <hip/hip_runtime.h>
#include <hip/hip_bf16.h>

// ---------- CDNA5 WMMA / TDM types ----------
typedef __attribute__((ext_vector_type(16))) __bf16 v16bf;
typedef __attribute__((ext_vector_type(8)))  float  v8f;
typedef __attribute__((ext_vector_type(4)))  unsigned int u32x4;
typedef __attribute__((ext_vector_type(8)))  unsigned int u32x8;

union BF16Frag {
    v16bf v;
    unsigned int   u32[8];
    unsigned short u[16];
};

// fp32 -> bf16 RNE (used once in the weight-conversion pass)
__device__ __forceinline__ unsigned short f2bf_rne(float f) {
    union { float f; unsigned int u; } x;
    x.f = f;
    unsigned int u = x.u;
    unsigned int r = u + 0x7FFFu + ((u >> 16) & 1u);
    return (unsigned short)(r >> 16);
}
// fp32 -> bf16 truncation (hot path: 1 VALU op)
__device__ __forceinline__ unsigned short f2bf_t(float f) {
    union { float f; unsigned int u; } x;
    x.f = f;
    return (unsigned short)(x.u >> 16);
}
// pack two fp32 into one u32 of bf16 pair (2 VALU ops)
__device__ __forceinline__ unsigned int pack2bf_t(float a, float b) {
    union { float f; unsigned int u; } ua, ub;
    ua.f = a; ub.f = b;
    return (ua.u >> 16) | (ub.u & 0xFFFF0000u);
}

#define LOG2_127F 6.9886846867721655f

// ReLU + integer-step quantize + log2 bit truncation; division-free, branchless
__device__ __forceinline__ float int_relu_f(float v, float divq, float rdiv) {
    v = fmaxf(v, 0.0f);
    float q  = truncf(v * rdiv);
    float lg = log2f(fmaxf(q, 2.0f));
    float tt = truncf(lg - LOG2_127F);
    float qb = truncf(q * exp2f(-tt)) * exp2f(tt);
    return ((q > 127.0f) ? qb : q) * divq;
}

// =====================================================================
// One-time (per call) conv weight convert + zero-pad to bf16 [Cout][Kpad]
// =====================================================================
__global__ void cvt_pad_weights(const float* __restrict__ w,
                                unsigned short* __restrict__ o,
                                int K, int Kpad)
{
    int k  = blockIdx.x * blockDim.x + threadIdx.x;
    int co = blockIdx.y;
    if (k >= Kpad) return;
    unsigned short v = 0;
    if (k < K) v = f2bf_rne(w[(size_t)co * K + k]);
    o[(size_t)co * Kpad + k] = v;
}

// =====================================================================
// Implicit-GEMM 3x3 conv (pad=1, stride=1), bf16 WMMA, fp32 accumulate.
// Block = 256 threads = 8 waves. Block tile: 64 (Cout) x 32 (pixels).
// Pixel tile = 32 contiguous x within one (b, y) row -> branchless im2col.
// K-step = 64: two WMMAs per wave per barrier pair.
// A (weights, bf16 pre-padded) tile moved by the Tensor Data Mover:
//   wave0 issues tensor_load_to_lds (64x64 bf16 2D tile, row stride Kpad),
//   overlapped with the im2col gather, drained with s_wait_tensorcnt.
// =====================================================================
__global__ __launch_bounds__(256) void conv3x3_wmma_bf16(
    const float* __restrict__ in,            // [B][Cin][H][W]
    const unsigned short* __restrict__ wgt,  // bf16 [Cout][Kpad]
    const float* __restrict__ bias,          // [Cout]
    const float* __restrict__ peek,
    int li,
    float* __restrict__ out,                 // [B][Cout][H][W]
    int Cin, int H, int W, int Cout, int xTiles, int Kpad)
{
    const int K = Cin * 9;

    const int tid  = threadIdx.x;
    const int wave = tid >> 5;
    const int lane = tid & 31;
    const int msub = wave & 3;   // 0..3 (M subtile)
    const int nsub = wave >> 2;  // 0..1 (N subtile)
    const int half = lane >> 4;  // 0/1
    const int ml   = lane & 15;  // 0..15

    // block -> (b, y, xbase)
    int t = blockIdx.x;
    const int xbase = (t % xTiles) * 32;  t /= xTiles;
    const int y = t % H;
    const int b = t / H;
    const int mbase = blockIdx.y * 64;

    // LDS: bf16 pairs packed in u32.  A: [64 rows][32 pairs]  B: [64 k][16 pairs]
    __shared__ __align__(16) unsigned int Alds32[64 * 32];   // 8 KB, TDM destination
    __shared__ __align__(16) unsigned int Blds32[64 * 16];   // 4 KB

    v8f acc = {0.f, 0.f, 0.f, 0.f, 0.f, 0.f, 0.f, 0.f};

    // staging roles for B (constant per thread)
    const int bkk  = tid >> 2;         // 0..63  (B k row)
    const int bnp  = (tid & 3) * 4;    // pair offset 0,4,8,12  (8 n values)

    // TDM D# invariants (uniform): lds dst, tile geometry, tensor dims/strides
    const unsigned ldsAddr = (unsigned)(size_t)&Alds32[0];
    const unsigned short* wrow0 = wgt + (size_t)mbase * Kpad;

    for (int k0 = 0; k0 < Kpad; k0 += 64) {
        // ---- wave0: kick TDM for the A tile (64 rows x 64 bf16, stride Kpad)
        if (wave == 0) {
            unsigned long long ga =
                (unsigned long long)(size_t)(wrow0 + k0);   // tile start, bytes
            u32x4 g0;
            g0.x = 1u;                                      // count=1, user mode
            g0.y = ldsAddr;                                 // D#.lds_addr
            g0.z = (unsigned)(ga & 0xFFFFFFFFu);            // global_addr[31:0]
            g0.w = (unsigned)((ga >> 32) & 0x01FFFFFFu)     // global_addr[56:32]
                 | (2u << 30);                              // type = 2 ("image")
            u32x8 g1;
            g1.s0 = (1u << 16);                             // data_size=1 -> 2 bytes
            g1.s1 = ((unsigned)Kpad & 0xFFFFu) << 16;       // tensor_dim0[15:0]
            g1.s2 = ((unsigned)Kpad >> 16)                  // tensor_dim0[31:16]
                  | (((unsigned)Cout & 0xFFFFu) << 16);     // tensor_dim1[15:0]
            g1.s3 = (64u << 16);                            // tile_dim0 = 64
            g1.s4 = 64u;                                    // tile_dim1=64, tile_dim2=0
            g1.s5 = (unsigned)Kpad;                         // tensor_dim0_stride[31:0]
            g1.s6 = 0u;                                     // stride hi / dim1_stride
            g1.s7 = 0u;
            asm volatile("tensor_load_to_lds %0, %1"
                         :: "s"(g0), "s"(g1)
                         : "memory");
        }

        // ---- stage B (im2col): one k-row x 8 n per thread, branchless
        {
            int k  = k0 + bkk;
            int kc = min(k, K - 1);
            int ci = kc / 9, r = kc % 9;
            int ky = r / 3,  kx = r % 3;
            int iy = y + ky - 1;
            bool rowOK = (k < K) && ((unsigned)iy < (unsigned)H);
            int iyc = min(max(iy, 0), H - 1);
            const size_t base = (((size_t)b * Cin + ci) * H + iyc) * W;
            #pragma unroll
            for (int i2 = 0; i2 < 4; ++i2) {
                int x0 = xbase + bnp * 2 + 2 * i2 + kx - 1;
                int x1 = x0 + 1;
                float v0 = in[base + min(max(x0, 0), W - 1)];
                float v1 = in[base + min(max(x1, 0), W - 1)];
                v0 = (rowOK && (unsigned)x0 < (unsigned)W) ? v0 : 0.0f;
                v1 = (rowOK && (unsigned)x1 < (unsigned)W) ? v1 : 0.0f;
                Blds32[bkk * 16 + bnp + i2] = pack2bf_t(v0, v1);
            }
        }

        // ---- drain TDM on the issuing wave, then rendezvous
        if (wave == 0)
            __builtin_amdgcn_s_wait_tensorcnt(0);
        __syncthreads();

        // ---- two 16x16x32 WMMAs (K halves 0..31, 32..63)
        const int arowi = msub * 16 + ml;
        #pragma unroll
        for (int h2 = 0; h2 < 2; ++h2) {
            BF16Frag af, bf;
            #pragma unroll
            for (int j = 0; j < 8; ++j) {
                // A pair index within 32-k window (ISA 16-bit A 16x32 layout)
                int pidx = (j < 4) ? (j + 4 * half) : (8 + (j - 4) + 4 * half);
                af.u32[j] = Alds32[arowi * 32 + h2 * 16 + pidx];
            }
            #pragma unroll
            for (int s = 0; s < 16; ++s) {  // lane = N, components = K (table order)
                int kk = (s < 8) ? (8 * half + s) : (16 + 8 * half + (s - 8));
                bf.u[s] = ((const unsigned short*)Blds32)[(h2 * 32 + kk) * 32 + nsub * 16 + ml];
            }
            acc = __builtin_amdgcn_wmma_f32_16x16x32_bf16(
                      false, af.v, false, bf.v, (short)0, acc, false, false);
        }
        __syncthreads();
    }

    // ---- epilogue: lane -> n = lane&15, VGPR r -> m = r + 8*half
    const float divq = peek[li];
    const float rdiv = __builtin_amdgcn_rcpf(divq);
    const int x = xbase + nsub * 16 + (lane & 15);
    if (x < W) {
        #pragma unroll
        for (int r = 0; r < 8; ++r) {
            int co  = mbase + msub * 16 + r + 8 * half;
            float v = acc[r] + bias[co];
            v = int_relu_f(v, divq, rdiv);
            out[(((size_t)b * Cout + co) * H + y) * W + x] = v;
        }
    }
}

// =====================================================================
// FC layer: out[n][m] = sum_k x[n][k]*w[m][k] + b[m]; WMMA 16x16 tiles,
// batch 8 padded to 16. One wave per 16-row M tile, float2 A loads.
// =====================================================================
__global__ __launch_bounds__(256) void fc_wmma_bf16(
    const float* __restrict__ x,     // [8][Din]
    const float* __restrict__ wgt,   // [Dout][Din]
    const float* __restrict__ bias,  // [Dout]
    const float* __restrict__ peek,
    int li, int doRelu,
    float* __restrict__ out,         // [8][Dout]
    int Din, int Dout)
{
    const int wave  = threadIdx.x >> 5;
    const int lane  = threadIdx.x & 31;
    const int gw    = blockIdx.x * 8 + wave;
    const int mbase = gw * 16;
    if (mbase >= Dout) return;  // wave-uniform exit

    const int half = lane >> 4;
    const int ml   = lane & 15;
    const int mrow = mbase + ml;
    const int mr   = (mrow < Dout) ? mrow : (Dout - 1);  // clamp; bad rows never stored

    const float2* wg2 = (const float2*)wgt;              // Din is even for all layers
    const size_t  wrow2 = (size_t)mr * (Din >> 1);

    v8f acc = {0.f, 0.f, 0.f, 0.f, 0.f, 0.f, 0.f, 0.f};

    for (int k0 = 0; k0 < Din; k0 += 32) {   // Din % 32 == 0 for all FC layers
        BF16Frag af, bf;
        #pragma unroll
        for (int j = 0; j < 8; ++j) {
            int kb = ((j < 4) ? (2 * j) : (16 + 2 * (j - 4))) + 8 * half;
            float2 w = wg2[wrow2 + ((k0 + kb) >> 1)];
            af.u32[j] = pack2bf_t(w.x, w.y);
        }
        const int nB = lane & 15;            // B fragment: lane = N, elements = K
        #pragma unroll
        for (int s = 0; s < 16; ++s) {       // batch == 8; cols 8..15 zero
            int kk = (s < 8) ? (8 * half + s) : (16 + 8 * half + (s - 8));
            float xv = x[(size_t)min(nB, 7) * Din + k0 + kk];
            bf.u[s] = (nB < 8) ? f2bf_t(xv) : (unsigned short)0;
        }
        acc = __builtin_amdgcn_wmma_f32_16x16x32_bf16(
                  false, af.v, false, bf.v, (short)0, acc, false, false);
    }

    const float divq = peek[li];
    const float rdiv = __builtin_amdgcn_rcpf(divq);
    const int n = lane & 15;
    if (n < 8) {
        #pragma unroll
        for (int r = 0; r < 8; ++r) {
            int m = mbase + r + 8 * half;
            if (m < Dout) {
                float v = acc[r] + bias[m];
                if (doRelu) v = int_relu_f(v, divq, rdiv);
                out[(size_t)n * Dout + m] = v;
            }
        }
    }
}

// =====================================================================
// Max pool (generic k/s/p, -inf padding) and exact 4x4 mean pool (28->7)
// =====================================================================
__global__ void maxpool_k(const float* __restrict__ in, float* __restrict__ out,
                          int B, int C, int H, int W, int k, int s, int p,
                          int Ho, int Wo)
{
    int idx = blockIdx.x * blockDim.x + threadIdx.x;
    int total = B * C * Ho * Wo;
    if (idx >= total) return;
    int wo = idx % Wo; int t = idx / Wo;
    int ho = t % Ho;   t /= Ho;
    int c  = t % C;    int b = t / C;
    float m = -INFINITY;
    for (int dy = 0; dy < k; ++dy)
        for (int dx = 0; dx < k; ++dx) {
            int yy = ho * s - p + dy, xx = wo * s - p + dx;
            if ((unsigned)yy < (unsigned)H && (unsigned)xx < (unsigned)W)
                m = fmaxf(m, in[(((size_t)b * C + c) * H + yy) * W + xx]);
        }
    out[idx] = m;
}

__global__ void avgpool4_k(const float* __restrict__ in, float* __restrict__ out,
                           int B, int C)
{
    int idx = blockIdx.x * blockDim.x + threadIdx.x;
    int total = B * C * 49;
    if (idx >= total) return;
    int wo = idx % 7; int t = idx / 7;
    int ho = t % 7;   t /= 7;
    int c  = t % C;   int b = t / C;
    float s = 0.0f;
    for (int dy = 0; dy < 4; ++dy)
        for (int dx = 0; dx < 4; ++dx)
            s += in[(((size_t)b * C + c) * 28 + (ho * 4 + dy)) * 28 + (wo * 4 + dx)];
    out[idx] = s * 0.0625f;
}

// =====================================================================
// Host-side pipeline
// =====================================================================
extern "C" void kernel_launch(void* const* d_in, const int* in_sizes, int n_in,
                              void* d_out, int out_size, void* d_ws, size_t ws_size,
                              hipStream_t stream) {
    (void)in_sizes; (void)n_in; (void)out_size; (void)ws_size;

    const float* x    = (const float*)d_in[0];
    const float* peek = (const float*)d_in[1];
    const float* convw[13]; const float* convb[13];
    for (int i = 0; i < 13; ++i) {
        convw[i] = (const float*)d_in[2 + i];
        convb[i] = (const float*)d_in[15 + i];
    }
    const float* fcw[3]; const float* fcb[3];
    for (int i = 0; i < 3; ++i) {
        fcw[i] = (const float*)d_in[28 + i];
        fcb[i] = (const float*)d_in[31 + i];
    }

    const int chIn[13]  = {3, 64, 64, 128, 128, 256, 256, 256, 512, 512, 512, 512, 512};
    const int chOut[13] = {64, 64, 128, 128, 256, 256, 256, 512, 512, 512, 512, 512, 512};

    // workspace layout: 2 activation ping-pong buffers + bf16 padded weights
    const size_t maxAct = (size_t)8 * 64 * 224 * 224;  // elems
    float* bufs[2] = { (float*)d_ws, (float*)d_ws + maxAct };
    unsigned short* wbf = (unsigned short*)((float*)d_ws + 2 * maxAct);

    // ---- convert + pad all conv weights to bf16 [Cout][Kpad]
    unsigned short* wptr[13];
    {
        unsigned short* p = wbf;
        for (int i = 0; i < 13; ++i) {
            int K    = chIn[i] * 9;
            int Kpad = (K + 63) & ~63;
            wptr[i] = p;
            dim3 grid((Kpad + 255) / 256, chOut[i]);
            cvt_pad_weights<<<grid, 256, 0, stream>>>(convw[i], p, K, Kpad);
            p += (size_t)chOut[i] * Kpad;
        }
    }

    const int nConv[5]  = {2, 2, 3, 3, 3};
    const int pk[5] = {2, 2, 2, 2, 2};
    const int ps[5] = {2, 2, 2, 1, 1};
    const int pp[5] = {0, 0, 0, 1, 0};

    const int Bn = 8;
    int H = 224, W = 224;
    const float* cur = x;
    int pb = 0;
    int li = 0;

    for (int bi = 0; bi < 5; ++bi) {
        for (int ci = 0; ci < nConv[bi]; ++ci) {
            int Cin = chIn[li], Cout = chOut[li];
            int K    = Cin * 9;
            int Kpad = (K + 63) & ~63;
            int xTiles = (W + 31) / 32;
            dim3 grid(xTiles * H * Bn, Cout / 64);
            float* nxt = bufs[pb];
            conv3x3_wmma_bf16<<<grid, 256, 0, stream>>>(
                cur, wptr[li], convb[li], peek, li, nxt, Cin, H, W, Cout, xTiles, Kpad);
            cur = nxt; pb ^= 1;
            ++li;
        }
        int k = pk[bi], s = ps[bi], p = pp[bi];
        int C = chOut[li - 1];
        int Ho = (H + 2 * p - k) / s + 1;
        int Wo = (W + 2 * p - k) / s + 1;
        int total = Bn * C * Ho * Wo;
        float* nxt = bufs[pb];
        maxpool_k<<<(total + 255) / 256, 256, 0, stream>>>(
            cur, nxt, Bn, C, H, W, k, s, p, Ho, Wo);
        cur = nxt; pb ^= 1;
        H = Ho; W = Wo;
    }

    // Adaptive avg pool: 28x28 -> 7x7 (exact 4x4 mean)
    {
        int C = 512, total = Bn * C * 49;
        float* nxt = bufs[pb];
        avgpool4_k<<<(total + 255) / 256, 256, 0, stream>>>(cur, nxt, Bn, C);
        cur = nxt; pb ^= 1;
    }

    // FC1: 25088 -> 4096 (+int_relu, peek[13])
    {
        int Din = 512 * 49, Dout = 4096;
        float* nxt = bufs[pb];
        fc_wmma_bf16<<<(Dout / 16 + 7) / 8, 256, 0, stream>>>(
            cur, fcw[0], fcb[0], peek, 13, 1, nxt, Din, Dout);
        cur = nxt; pb ^= 1;
    }
    // FC2: 4096 -> 4096 (+int_relu, peek[14])
    {
        int Din = 4096, Dout = 4096;
        float* nxt = bufs[pb];
        fc_wmma_bf16<<<(Dout / 16 + 7) / 8, 256, 0, stream>>>(
            cur, fcw[1], fcb[1], peek, 14, 1, nxt, Din, Dout);
        cur = nxt; pb ^= 1;
    }
    // FC3: 4096 -> 1000, no relu, straight to d_out
    {
        int Din = 4096, Dout = 1000;
        fc_wmma_bf16<<<((Dout + 15) / 16 + 7) / 8, 256, 0, stream>>>(
            cur, fcw[2], fcb[2], peek, 14, 0, (float*)d_out, Din, Dout);
    }
}
